// Step1Attention_79766132621596
// MI455X (gfx1250) — compile-verified, matrix operand order checked by reading, not run on hardware
//
#include <hip/hip_runtime.h>
#include <hip/hip_bf16.h>

// ---------------------------------------------------------------------------
// Problem constants (from reference): B=2, S=2048, Hd=2048, H=16, G=4, D=128
// ---------------------------------------------------------------------------
#define H_   16
#define G_   4
#define REP  4          // H/G
#define B_   2
#define S_   2048
#define HD   2048
#define D_   128
#define MTOT (B_ * S_)  // 4096 token rows
#define KD   (G_ * D_)  // 512 = K/V projection width

typedef __attribute__((ext_vector_type(16))) __bf16 v16bf;
typedef __attribute__((ext_vector_type(8)))  float  v8f;
typedef __attribute__((ext_vector_type(4)))  unsigned int v4u;
typedef __attribute__((ext_vector_type(8)))  int v8i;
typedef __attribute__((ext_vector_type(4)))  int v4i;
typedef unsigned long long ull;

#if defined(__HIP_DEVICE_COMPILE__) && \
    __has_builtin(__builtin_amdgcn_tensor_load_to_lds) && \
    __has_builtin(__builtin_amdgcn_s_wait_tensorcnt)
#define HAVE_TDM 1
#else
#define HAVE_TDM 0
#endif

__device__ __forceinline__ unsigned short f2bf(float f) {
    unsigned u = __builtin_bit_cast(unsigned, f);
    u += 0x7FFFu + ((u >> 16) & 1u);           // round-to-nearest-even
    return (unsigned short)(u >> 16);
}

__device__ __forceinline__ v8f zero8() {
    v8f z;
    #pragma unroll
    for (int e = 0; e < 8; ++e) z[e] = 0.0f;
    return z;
}

#if HAVE_TDM
// Low 32 bits of a flat LDS-aperture address are the LDS byte offset
// (ISA 10.2).
__device__ __forceinline__ unsigned lds_off_u32(const void* p) {
    return (unsigned)(ull)p;
}

// Build a 2-D Tensor DMA descriptor (ISA ch.8) and issue TENSOR_LOAD_TO_LDS.
// data_size = 2 bytes (bf16). Optional LDS padding: after every
// 2^(pad_int_code+1) DWORDs insert (pad_amt_code+1) DWORDs.
// This toolchain exposes the 6-arg builtin:
//   (uint32x4 g0, int32x8 g1, int32x4 g2, int32x4 g3, int32x8 g4, i32 cpol)
__device__ __forceinline__ void tdm_load_2d(
    unsigned lds_byte, ull gaddr,
    unsigned tdim0, unsigned tdim1,          // tensor extents (OOB guard), elems
    unsigned tile0, unsigned tile1,          // tile extents, elems
    ull stride0,                             // tensor dim0 stride, elems
    unsigned pad_int_code, unsigned pad_amt_code, bool pad_en)
{
    v4u g0;
    g0[0] = 1u;                                                    // count=1, user D#
    g0[1] = lds_byte;                                              // lds_addr
    g0[2] = (unsigned)gaddr;                                       // global_addr[31:0]
    g0[3] = (unsigned)((gaddr >> 32) & 0x01FFFFFFu) | (2u << 30);  // [56:32] | type=2
    v8i g1;
    g1[0] = (int)((1u << 16) |                                     // data_size = 2B
                  (pad_en ? (1u << 20) : 0u) |
                  (pad_int_code << 22) | (pad_amt_code << 25));
    g1[1] = (int)((tdim0 & 0xFFFFu) << 16);                        // tensor_dim0[15:0]
    g1[2] = (int)((tdim0 >> 16) | ((tdim1 & 0xFFFFu) << 16));      // dim0 hi | dim1 lo
    g1[3] = (int)((tdim1 >> 16) | (tile0 << 16));                  // dim1 hi | tile_dim0
    g1[4] = (int)(tile1 & 0xFFFFu);                                // tile_dim1 (tile_dim2=0)
    g1[5] = (int)(unsigned)stride0;                                // dim0_stride lo32
    g1[6] = (int)((stride0 >> 32) & 0xFFFFu);                      // dim0_stride hi16
    g1[7] = 0;
    v4i z4 = {0, 0, 0, 0};
    v8i z8 = {0, 0, 0, 0, 0, 0, 0, 0};
    __builtin_amdgcn_tensor_load_to_lds(g0, g1, z4, z4, z8, 0);
}
#endif

// ---------------------------------------------------------------------------
// f32 -> bf16 conversion (one pass; trivial vs. 23.3 TB/s HBM)
// ---------------------------------------------------------------------------
__global__ __launch_bounds__(256) void cvt_f32_bf16(
    const float* __restrict__ src, unsigned short* __restrict__ dst, int n)
{
    int i = (blockIdx.x * 256 + threadIdx.x) * 4;
    if (i + 3 < n) {
        float4 f = *(const float4*)(src + i);
        ushort4 o;
        o.x = f2bf(f.x); o.y = f2bf(f.y); o.z = f2bf(f.z); o.w = f2bf(f.w);
        *(ushort4*)(dst + i) = o;
    }
}

// ---------------------------------------------------------------------------
// NT GEMM: C[m,n] = sum_k A[m,k] * B[n,k]  (A: MxK bf16, B: NxK bf16)
// Block tile 128x128, 8 waves in 4(M)x2(N), each wave 2x4 WMMA tiles.
// K staged in 32-wide chunks; TDM double-buffered (waves only do WMMA).
// ---------------------------------------------------------------------------
template <typename OutT>
__global__ __launch_bounds__(256) void gemm_nt(
    const unsigned short* __restrict__ A,
    const unsigned short* __restrict__ Bm,
    OutT* __restrict__ C, int M, int N, int K)
{
    __shared__ __attribute__((aligned(32))) unsigned short ldsA[2 * 128 * 48];
    __shared__ __attribute__((aligned(32))) unsigned short ldsB[2 * 128 * 48];

    const int tid  = threadIdx.x;
    const int lane = tid & 31;
    const int w    = tid >> 5;
    const int wm   = w & 3;          // 32-row slice
    const int wn   = w >> 2;         // 64-col slice
    const int half = lane >> 4;      // K-half select (k = 16*half + s)
    const int l16  = lane & 15;

    const int m0 = blockIdx.y * 128;
    const int n0 = blockIdx.x * 128;
    const int nt = K >> 5;

    v8f acc[2][4];
    #pragma unroll
    for (int i = 0; i < 2; ++i)
        #pragma unroll
        for (int j = 0; j < 4; ++j) acc[i][j] = zero8();

#if HAVE_TDM
    if (w == 0) {   // prologue: DMA tile 0 into buffer 0
        tdm_load_2d(lds_off_u32(ldsA), (ull)(A + (size_t)m0 * K),
                    (unsigned)K, (unsigned)(M - m0), 32, 128, (ull)K, 3, 7, true);
        tdm_load_2d(lds_off_u32(ldsB), (ull)(Bm + (size_t)n0 * K),
                    (unsigned)K, (unsigned)(N - n0), 32, 128, (ull)K, 3, 7, true);
    }
#endif

    for (int t = 0; t < nt; ++t) {
        const int kb = t << 5;
        unsigned short* LA = ldsA + (t & 1) * (128 * 48);
        unsigned short* LB = ldsB + (t & 1) * (128 * 48);

#if HAVE_TDM
        if (w == 0) {
            if (t + 1 < nt) {   // DMA next tile into the other buffer
                const int kb2 = kb + 32;
                const unsigned nb = (unsigned)(((t + 1) & 1) * (128 * 48 * 2));
                tdm_load_2d(lds_off_u32(ldsA) + nb,
                            (ull)(A + (size_t)m0 * K + kb2),
                            (unsigned)(K - kb2), (unsigned)(M - m0),
                            32, 128, (ull)K, 3, 7, true);
                tdm_load_2d(lds_off_u32(ldsB) + nb,
                            (ull)(Bm + (size_t)n0 * K + kb2),
                            (unsigned)(K - kb2), (unsigned)(N - n0),
                            32, 128, (ull)K, 3, 7, true);
                __builtin_amdgcn_s_wait_tensorcnt(2);   // pair t complete (in-order)
            } else {
                __builtin_amdgcn_s_wait_tensorcnt(0);
            }
        }
        __syncthreads();
#else
        // Fallback / host pass: cooperative manual staging.
        #pragma unroll
        for (int j = 0; j < 2; ++j) {
            int idx = tid * 2 + j;               // 0..511
            int row = idx >> 2;
            int c8  = (idx & 3) * 8;
            *(uint4*)(LA + row * 48 + c8) =
                *(const uint4*)(A + (size_t)(m0 + row) * K + kb + c8);
            *(uint4*)(LB + row * 48 + c8) =
                *(const uint4*)(Bm + (size_t)(n0 + row) * K + kb + c8);
        }
        __syncthreads();
#endif

        v16bf af[2], bf[4];
        #pragma unroll
        for (int i = 0; i < 2; ++i)
            af[i] = *(const v16bf*)(LA + (wm * 32 + i * 16 + l16) * 48 + half * 16);
        #pragma unroll
        for (int j = 0; j < 4; ++j)
            bf[j] = *(const v16bf*)(LB + (wn * 64 + j * 16 + l16) * 48 + half * 16);

        #pragma unroll
        for (int i = 0; i < 2; ++i)
            #pragma unroll
            for (int j = 0; j < 4; ++j)
                acc[i][j] = __builtin_amdgcn_wmma_f32_16x16x32_bf16(
                    false, af[i], false, bf[j], (short)0, acc[i][j], false, false);
        __syncthreads();   // reads done before buffer reused
    }

    // C-tile layout (ISA 7.12.2): m = vgpr + 8*half, n = lane%16
    #pragma unroll
    for (int i = 0; i < 2; ++i)
        #pragma unroll
        for (int j = 0; j < 4; ++j)
            #pragma unroll
            for (int r = 0; r < 8; ++r) {
                int m = m0 + wm * 32 + i * 16 + r + half * 8;
                int n = n0 + wn * 64 + j * 16 + l16;
                float val = acc[i][j][r];
                if constexpr (sizeof(OutT) == 2)
                    ((unsigned short*)C)[(size_t)m * N + n] = f2bf(val);
                else
                    ((float*)C)[(size_t)m * N + n] = val;
            }
}

// ---------------------------------------------------------------------------
// Flash-style GQA attention with ALiBi-sqrt bias, causal.
// Grid: (S/128, B*H). 256 threads = 8 waves; wave owns 16 q-rows.
// K tile (32x128) staged by TDM; V staged transposed so P@V is an NT WMMA.
// ---------------------------------------------------------------------------
__global__ __launch_bounds__(256) void attn_kernel(
    const unsigned short* __restrict__ qb,
    const unsigned short* __restrict__ kbuf,
    const unsigned short* __restrict__ vbuf,
    unsigned short* __restrict__ ob)
{
    __shared__ __attribute__((aligned(32))) unsigned short ldsK[32 * 144];  // [key][d], TDM-padded
    __shared__ __attribute__((aligned(32))) unsigned short ldsV[128 * 48];  // Vt[d][kk]
    __shared__ __attribute__((aligned(32))) unsigned short ldsP[8 * 16 * 32];

    const int tid  = threadIdx.x;
    const int lane = tid & 31;
    const int w    = tid >> 5;
    const int half = lane >> 4;
    const int l16  = lane & 15;

    const int qblk = blockIdx.x;            // 0..15
    const int bh   = blockIdx.y;            // b*H + h
    const int h    = bh % H_;
    const int b    = bh / H_;
    const int g    = h / REP;               // K/V group
    const int qrow0 = qblk * 128 + w * 16;

    const float scale = 0.08838834764831845f;           // 1/sqrt(128)
    const float slope = exp2f(-0.5f * (float)(h + 1));  // ALiBi slope

    // q fragments for this wave's 16 rows (K=128 -> 4 WMMA steps), in VGPRs
    v16bf qf[4];
    {
        const unsigned short* qp =
            qb + ((size_t)(b * S_ + qrow0 + l16) * HD + h * D_);
        #pragma unroll
        for (int ks = 0; ks < 4; ++ks)
            qf[ks] = *(const v16bf*)(qp + ks * 32 + half * 16);
    }

    v8f oacc[8];
    #pragma unroll
    for (int dt = 0; dt < 8; ++dt) oacc[dt] = zero8();
    float mi[8], li[8];
    #pragma unroll
    for (int r = 0; r < 8; ++r) { mi[r] = -__builtin_inff(); li[r] = 0.0f; }

    const int kend = qblk * 128 + 128;      // causal bound (uniform per block)
    for (int kb = 0; kb < kend; kb += 32) {
#if HAVE_TDM
        if (w == 0) {   // DMA K tile [32 keys][128 d] with pad -> 144 stride
            tdm_load_2d(lds_off_u32(ldsK),
                        (ull)(kbuf + (size_t)(b * S_ + kb) * KD + g * D_),
                        (unsigned)(KD - g * D_), (unsigned)(MTOT - (b * S_ + kb)),
                        128, 32, (ull)KD, 5, 7, true);
        }
#endif
        // V tile staged transposed: Vt[d][kk] (TDM cannot transpose)
        #pragma unroll
        for (int j = 0; j < 2; ++j) {
            int idx = tid * 2 + j;           // 0..511
            int row = idx >> 4;              // key in tile
            int c8  = (idx & 15) * 8;        // d offset
#if !HAVE_TDM
            uint4 kd4 = *(const uint4*)(kbuf + (size_t)(b * S_ + kb + row) * KD + g * D_ + c8);
            *(uint4*)(ldsK + row * 144 + c8) = kd4;
#endif
            uint4 vd = *(const uint4*)(vbuf + (size_t)(b * S_ + kb + row) * KD + g * D_ + c8);
            const unsigned short* ve = (const unsigned short*)&vd;
            #pragma unroll
            for (int e = 0; e < 8; ++e)
                ldsV[(c8 + e) * 48 + row] = ve[e];
        }
#if HAVE_TDM
        if (w == 0) __builtin_amdgcn_s_wait_tensorcnt(0);
#endif
        __syncthreads();

        // S = q . k^T : two 16x16 key tiles, K=128 in 4 WMMA steps
        v8f sacc[2];
        sacc[0] = zero8(); sacc[1] = zero8();
        #pragma unroll
        for (int kt = 0; kt < 2; ++kt)
            #pragma unroll
            for (int ks = 0; ks < 4; ++ks) {
                v16bf kf = *(const v16bf*)(ldsK + (kt * 16 + l16) * 144 + ks * 32 + half * 16);
                sacc[kt] = __builtin_amdgcn_wmma_f32_16x16x32_bf16(
                    false, qf[ks], false, kf, (short)0, sacc[kt], false, false);
            }

        // ALiBi bias + causal mask + online softmax. Row m = r + 8*half lives
        // in one 16-lane half -> shfl_xor masks 1..8 stay inside the half.
        float p0[8], p1[8];
        #pragma unroll
        for (int r = 0; r < 8; ++r) {
            int qi = qrow0 + r + half * 8;
            int j0 = kb + l16;
            int j1 = kb + 16 + l16;
            float v0 = (j0 <= qi) ? (sacc[0][r] * scale - slope * sqrtf((float)(qi - j0)))
                                  : -__builtin_inff();
            float v1 = (j1 <= qi) ? (sacc[1][r] * scale - slope * sqrtf((float)(qi - j1)))
                                  : -__builtin_inff();
            float rm = fmaxf(v0, v1);
            #pragma unroll
            for (int s = 8; s >= 1; s >>= 1) rm = fmaxf(rm, __shfl_xor(rm, s, 32));
            float mn = fmaxf(mi[r], rm);
            float e0 = __expf(v0 - mn);
            float e1 = __expf(v1 - mn);
            float rs = e0 + e1;
            #pragma unroll
            for (int s = 8; s >= 1; s >>= 1) rs += __shfl_xor(rs, s, 32);
            float alpha = __expf(mi[r] - mn);
            li[r] = li[r] * alpha + rs;
            mi[r] = mn;
            p0[r] = e0; p1[r] = e1;
            #pragma unroll
            for (int dt = 0; dt < 8; ++dt) oacc[dt][r] *= alpha;
        }

        // C-layout -> A-fragment layout through per-wave LDS slot (16x32 bf16)
        #pragma unroll
        for (int r = 0; r < 8; ++r) {
            int m = r + half * 8;
            ldsP[w * 512 + m * 32 + l16]      = f2bf(p0[r]);
            ldsP[w * 512 + m * 32 + 16 + l16] = f2bf(p1[r]);
        }
        v16bf pf = *(const v16bf*)(ldsP + w * 512 + l16 * 32 + half * 16);

        // O += P @ V : D=128 -> 8 N-tiles, B-fragments from transposed Vt
        #pragma unroll
        for (int dt = 0; dt < 8; ++dt) {
            v16bf vf = *(const v16bf*)(ldsV + (dt * 16 + l16) * 48 + half * 16);
            oacc[dt] = __builtin_amdgcn_wmma_f32_16x16x32_bf16(
                false, pf, false, vf, (short)0, oacc[dt], false, false);
        }
        __syncthreads();   // before restaging ldsK/ldsV
    }

    // epilogue: normalize by softmax denom, store bf16 into o[B,S,Hd]
    #pragma unroll
    for (int r = 0; r < 8; ++r) {
        float inv = 1.0f / li[r];
        int m = qrow0 + r + half * 8;
        unsigned short* op = ob + ((size_t)(b * S_ + m) * HD + h * D_);
        #pragma unroll
        for (int dt = 0; dt < 8; ++dt)
            op[dt * 16 + l16] = f2bf(oacc[dt][r] * inv);
    }
}

// ---------------------------------------------------------------------------
// Host-side orchestration
// ---------------------------------------------------------------------------
extern "C" void kernel_launch(void* const* d_in, const int* in_sizes, int n_in,
                              void* d_out, int out_size, void* d_ws, size_t ws_size,
                              hipStream_t stream) {
    (void)in_sizes; (void)n_in; (void)out_size; (void)ws_size;
    const float* x  = (const float*)d_in[0];
    const float* Wq = (const float*)d_in[1];
    const float* Wk = (const float*)d_in[2];
    const float* Wv = (const float*)d_in[3];
    const float* Wo = (const float*)d_in[4];

    char* ws = (char*)d_ws;
    size_t off = 0;
    auto carve = [&](size_t bytes) { void* p = ws + off; off += bytes; return p; };
    unsigned short* xb  = (unsigned short*)carve((size_t)MTOT * HD * 2);
    unsigned short* Wqb = (unsigned short*)carve((size_t)HD * HD * 2);
    unsigned short* Wkb = (unsigned short*)carve((size_t)KD * HD * 2);
    unsigned short* Wvb = (unsigned short*)carve((size_t)KD * HD * 2);
    unsigned short* Wob = (unsigned short*)carve((size_t)HD * HD * 2);
    unsigned short* qbf = (unsigned short*)carve((size_t)MTOT * HD * 2);
    unsigned short* kbf = (unsigned short*)carve((size_t)MTOT * KD * 2);
    unsigned short* vbf = (unsigned short*)carve((size_t)MTOT * KD * 2);
    unsigned short* obf = (unsigned short*)carve((size_t)MTOT * HD * 2);

    auto cvt = [&](const float* s, unsigned short* d, int n) {
        cvt_f32_bf16<<<(n / 4 + 255) / 256, 256, 0, stream>>>(s, d, n);
    };
    cvt(x,  xb,  MTOT * HD);
    cvt(Wq, Wqb, HD * HD);
    cvt(Wk, Wkb, KD * HD);
    cvt(Wv, Wvb, KD * HD);
    cvt(Wo, Wob, HD * HD);

    // q = x @ Wq^T ; k = x @ Wk^T ; v = x @ Wv^T  (NT GEMMs, bf16 out)
    gemm_nt<unsigned short><<<dim3(HD / 128, MTOT / 128), 256, 0, stream>>>(
        xb, Wqb, qbf, MTOT, HD, HD);
    gemm_nt<unsigned short><<<dim3(KD / 128, MTOT / 128), 256, 0, stream>>>(
        xb, Wkb, kbf, MTOT, KD, HD);
    gemm_nt<unsigned short><<<dim3(KD / 128, MTOT / 128), 256, 0, stream>>>(
        xb, Wvb, vbf, MTOT, KD, HD);

    // Flash attention with ALiBi, GQA
    attn_kernel<<<dim3(S_ / 128, B_ * H_), 256, 0, stream>>>(qbf, kbf, vbf, obf);

    // out = o @ Wo^T  (f32 out)
    gemm_nt<float><<<dim3(HD / 128, MTOT / 128), 256, 0, stream>>>(
        obf, Wob, (float*)d_out, MTOT, HD, HD);
}
